// LocalGNN_12910671692048
// MI455X (gfx1250) — matrix operand order repaired and checked by reading, not compiled, not run on hardware
//
#include <hip/hip_runtime.h>
#include <hip/hip_bf16.h>

typedef __bf16 bf16_t;
typedef __attribute__((ext_vector_type(16))) __bf16 v16bf;
typedef __attribute__((ext_vector_type(8)))  __bf16 v8bf;
typedef __attribute__((ext_vector_type(8)))  float  v8f;

// ---------------------------------------------------------------------------
// Z[M,N] = X[M,N] * S[N,N], X bf16, S fp32 (converted to bf16 in-flight),
// accumulate f32 with v_wmma_f32_16x16x32_bf16. Writes Zf (f32) and Zb (bf16).
// Workgroup = 128 threads (4 wave32) owns 64 columns x all M rows, so each S
// element is pulled from HBM exactly once per GEMM (S-stream bound by design).
// Double-buffered LDS staging; transpose staging packs two adjacent k-rows per
// thread so each column becomes one v_cvt_pk_bf16_f32 + one ds_store_b32.
// RT = number of 16-row tiles (1 for layer-1 M=16, 8 for layer-2 M=128).
// ---------------------------------------------------------------------------
template<int RT>
__global__ __launch_bounds__(128)
void gemm_xs_bf16(const bf16_t* __restrict__ X, const float* __restrict__ S,
                  float* __restrict__ Zf, bf16_t* __restrict__ Zb, int N)
{
    __shared__ __attribute__((aligned(64))) bf16_t sB[2][64 * 32];       // [col][k]
    __shared__ __attribute__((aligned(64))) bf16_t sA[2][RT * 16 * 32];  // [row][k]

    const int t      = threadIdx.x;
    const int wave   = t >> 5;
    const int lane   = t & 31;
    const int laneLo = lane & 15;
    const int laneHi = lane >> 4;           // 0 or 1
    const int colBase = blockIdx.x * 64;

    const int c4 = (t & 15) * 4;    // 4-column group for S staging
    const int k2 = (t >> 4) * 2;    // even k-row pair for S staging (0..14)

    constexpr int XC = (RT * 64 + 127) / 128;   // 16B X-chunks per thread

    const v8f vzero = {0.f, 0.f, 0.f, 0.f, 0.f, 0.f, 0.f, 0.f};
    v8f acc[RT];
#pragma unroll
    for (int r = 0; r < RT; ++r) acc[r] = vzero;

    float4 sreg[2][2];   // [pass][row-of-pair]
    uint4  xreg[XC];

    // Issue global loads for K-chunk k0 into registers (no LDS yet).
    auto globalLoad = [&](int k0) {
#pragma unroll
        for (int p = 0; p < 2; ++p) {
            const int kk = k2 + p * 16;
            sreg[p][0] = *reinterpret_cast<const float4*>(
                &S[(size_t)(k0 + kk + 0) * N + (colBase + c4)]);
            sreg[p][1] = *reinterpret_cast<const float4*>(
                &S[(size_t)(k0 + kk + 1) * N + (colBase + c4)]);
        }
#pragma unroll
        for (int q = 0; q < XC; ++q) {
            const int ci = t + q * 128;
            if (ci < RT * 64) {
                const int row = ci >> 2;
                const int off = (ci & 3) * 8;
                xreg[q] = *reinterpret_cast<const uint4*>(&X[(size_t)row * N + k0 + off]);
            }
        }
    };

    // Convert + store staged registers into LDS buffer `buf`.
    // For each column c: pack (S[k][c], S[k+1][c]) -> one bf16 pair -> b32 store.
    auto ldsStore = [&](int buf) {
#pragma unroll
        for (int p = 0; p < 2; ++p) {
            const int kk = k2 + p * 16;
            const float* fa = reinterpret_cast<const float*>(&sreg[p][0]);
            const float* fb = reinterpret_cast<const float*>(&sreg[p][1]);
#pragma unroll
            for (int c = 0; c < 4; ++c) {
                union { bf16_t h[2]; unsigned u; } pk;
                pk.h[0] = (bf16_t)fa[c];
                pk.h[1] = (bf16_t)fb[c];
                *reinterpret_cast<unsigned*>(&sB[buf][(c4 + c) * 32 + kk]) = pk.u;
            }
        }
#pragma unroll
        for (int q = 0; q < XC; ++q) {
            const int ci = t + q * 128;
            if (ci < RT * 64) {
                const int row = ci >> 2;
                const int off = (ci & 3) * 8;
                *reinterpret_cast<uint4*>(&sA[buf][row * 32 + off]) = xreg[q];
            }
        }
    };

    // 8 (or 1) WMMAs for the chunk resident in LDS buffer `buf`.
    auto compute = [&](int buf) {
        // B operand (contiguous k order, not the ISA table order): lane n<16 -> col n, K0..15;
        // lane n+16 -> col n, K16..31. Col-major sB makes this contiguous.
        const v16bf Bv = *reinterpret_cast<const v16bf*>(
            &sB[buf][(wave * 16 + laneLo) * 32 + laneHi * 16]);
#pragma unroll
        for (int r = 0; r < RT; ++r) {
            // A operand (matches B k order): lane L<16 -> row L, K 0..15;
            // lane L+16 -> row L, K 16..31
            const int arow  = r * 16 + laneLo;
            const int abase = laneHi * 16;
            union { v16bf v; v8bf h[2]; } A;
            A.h[0] = *reinterpret_cast<const v8bf*>(&sA[buf][arow * 32 + abase]);
            A.h[1] = *reinterpret_cast<const v8bf*>(&sA[buf][arow * 32 + abase + 8]);
            acc[r] = __builtin_amdgcn_wmma_f32_16x16x32_bf16(
                false, A.v, false, Bv, (short)0, acc[r], false, false);
        }
    };

    // Prologue: stage chunk 0.
    globalLoad(0);
    ldsStore(0);
    __syncthreads();

    int buf = 0;
    for (int k0 = 32; k0 < N; k0 += 32) {
        globalLoad(k0);     // HBM loads in flight ...
        compute(buf);       // ... while WMMAs run on the resident chunk
        ldsStore(buf ^ 1);  // safe: all waves finished reading buf^1 before last barrier
        __syncthreads();
        buf ^= 1;
    }
    compute(buf);           // epilogue chunk

    // Epilogue: C/D layout -> VGPR i: lanes 0-15 row i, lanes 16-31 row i+8
    const int colOut = colBase + wave * 16 + laneLo;
    const int rowAdd = laneHi * 8;
#pragma unroll
    for (int r = 0; r < RT; ++r) {
#pragma unroll
        for (int i = 0; i < 8; ++i) {
            const int row = r * 16 + rowAdd + i;
            const float v = acc[r][i];
            Zf[(size_t)row * N + colOut] = v;
            Zb[(size_t)row * N + colOut] = (bf16_t)v;
        }
    }
}

// Cast x [8,N] f32 -> Xa [16,N] bf16, rows 8..15 zero-padded.
__global__ void prep_x(const float* __restrict__ x, bf16_t* __restrict__ Xa, int N)
{
    const int n = blockIdx.x * blockDim.x + threadIdx.x;
    if (n >= N) return;
#pragma unroll
    for (int b = 0; b < 8; ++b) Xa[(size_t)b * N + n] = (bf16_t)x[(size_t)b * N + n];
#pragma unroll
    for (int r = 8; r < 16; ++r) Xa[(size_t)r * N + n] = (bf16_t)0.f;
}

// Layer-1 tap combine + bias + ReLU:
// y1[b,g,n] = relu(sum_k H1[g,k,0] * z_k[b,n] + b1[g]); z0 = x.
// Emits Y0f (f32, for exact layer-2 combine) and Y0b (bf16, layer-2 GEMM input),
// rows flattened as m = b*16+g.
__global__ void combine1(const float* __restrict__ x,  const float* __restrict__ Z1,
                         const float* __restrict__ Z2, const float* __restrict__ Z3,
                         const float* __restrict__ H1, const float* __restrict__ b1,
                         float* __restrict__ Y0f, bf16_t* __restrict__ Y0b, int N)
{
    const int idx = blockIdx.x * blockDim.x + threadIdx.x;  // over B*N
    if (idx >= 8 * N) return;
    const int b = idx / N;
    const int n = idx - b * N;
    float z[4];
    z[0] = x[idx]; z[1] = Z1[idx]; z[2] = Z2[idx]; z[3] = Z3[idx];  // row b of [16,N] buffers
#pragma unroll
    for (int g = 0; g < 16; ++g) {
        float a = b1[g];
#pragma unroll
        for (int k = 0; k < 4; ++k) a += H1[g * 4 + k] * z[k];  // H1[F1=16][K=4][F0=1]
        a = fmaxf(a, 0.f);
        const size_t o = ((size_t)b * 16 + g) * N + n;
        Y0f[o] = a;
        Y0b[o] = (bf16_t)a;
    }
}

// Layer-2 tap combine + bias + ReLU + per-node readout:
// u[g] = relu(sum_{k,f} H2[g,k,f] * z_k[b,f,n] + b2[g]);  z0 = Y0f, zk = Wk.
// out[b,r,n] = sum_g Wro[r,g] * u[g] + bro[r]
__global__ void combine2(const float* __restrict__ Y0f, const float* __restrict__ W1,
                         const float* __restrict__ W2,  const float* __restrict__ W3,
                         const float* __restrict__ H2,  const float* __restrict__ b2,
                         const float* __restrict__ Wro, const float* __restrict__ bro,
                         float* __restrict__ out, int N)
{
    const int idx = blockIdx.x * blockDim.x + threadIdx.x;  // over B*N
    if (idx >= 8 * N) return;
    const int b = idx / N;
    const int n = idx - b * N;

    float u[16];
#pragma unroll
    for (int g = 0; g < 16; ++g) u[g] = b2[g];

    const float* bufs[4] = {Y0f, W1, W2, W3};
#pragma unroll
    for (int k = 0; k < 4; ++k) {
        const float* __restrict__ Z = bufs[k];
#pragma unroll
        for (int f = 0; f < 16; ++f) {
            const float zv = Z[((size_t)b * 16 + f) * N + n];
#pragma unroll
            for (int g = 0; g < 16; ++g)
                u[g] += H2[(g * 4 + k) * 16 + f] * zv;   // H2[g][k][f]
        }
    }
#pragma unroll
    for (int g = 0; g < 16; ++g) u[g] = fmaxf(u[g], 0.f);

#pragma unroll
    for (int r = 0; r < 8; ++r) {
        float a = bro[r];
#pragma unroll
        for (int g = 0; g < 16; ++g) a += Wro[r * 16 + g] * u[g];
        out[((size_t)b * 8 + r) * N + n] = a;   // [B, R, N]
    }
}

extern "C" void kernel_launch(void* const* d_in, const int* in_sizes, int n_in,
                              void* d_out, int out_size, void* d_ws, size_t ws_size,
                              hipStream_t stream)
{
    (void)in_sizes; (void)n_in; (void)out_size; (void)ws_size;
    const int N = 16384;
    const int B = 8;

    const float* x   = (const float*)d_in[0];
    const float* S   = (const float*)d_in[1];
    const float* H1  = (const float*)d_in[2];
    const float* b1  = (const float*)d_in[3];
    const float* H2  = (const float*)d_in[4];
    const float* b2  = (const float*)d_in[5];
    const float* Wro = (const float*)d_in[6];
    const float* bro = (const float*)d_in[7];
    float* out = (float*)d_out;

    // Workspace layout (bytes), ~52 MB total
    char* ws = (char*)d_ws;
    bf16_t* Xa  = (bf16_t*)(ws + 0);          // [16,N] bf16   512K
    bf16_t* Xb  = (bf16_t*)(ws + 524288);     // [16,N] bf16   512K
    float*  Z1  = (float*) (ws + 1048576);    // [16,N] f32    1M
    float*  Z2  = (float*) (ws + 2097152);    // [16,N] f32    1M
    float*  Z3  = (float*) (ws + 3145728);    // [16,N] f32    1M
    bf16_t* Y0b = (bf16_t*)(ws + 4194304);    // [128,N] bf16  4M
    float*  Y0f = (float*) (ws + 8388608);    // [128,N] f32   8M
    float*  W1  = (float*) (ws + 16777216);   // [128,N] f32   8M
    float*  W2  = (float*) (ws + 25165824);   // [128,N] f32   8M
    float*  W3  = (float*) (ws + 33554432);   // [128,N] f32   8M
    bf16_t* XB1 = (bf16_t*)(ws + 41943040);   // [128,N] bf16  4M
    bf16_t* XB2 = (bf16_t*)(ws + 46137344);   // [128,N] bf16  4M
    bf16_t* XB3 = (bf16_t*)(ws + 50331648);   // [128,N] bf16  4M

    const dim3 gGemm(N / 64), bGemm(128);

    // Layer 1: z_k = x S^k (M=16 padded), then tap-combine + ReLU
    prep_x<<<(N + 255) / 256, 256, 0, stream>>>(x, Xa, N);
    gemm_xs_bf16<1><<<gGemm, bGemm, 0, stream>>>(Xa, S, Z1, Xb, N);
    gemm_xs_bf16<1><<<gGemm, bGemm, 0, stream>>>(Xb, S, Z2, Xa, N);
    gemm_xs_bf16<1><<<gGemm, bGemm, 0, stream>>>(Xa, S, Z3, Xb, N);
    combine1<<<(B * N + 255) / 256, 256, 0, stream>>>(x, Z1, Z2, Z3, H1, b1, Y0f, Y0b, N);

    // Layer 2: w_k = y1 S^k (M=128), then tap-combine + ReLU + readout
    gemm_xs_bf16<8><<<gGemm, bGemm, 0, stream>>>(Y0b, S, W1, XB1, N);
    gemm_xs_bf16<8><<<gGemm, bGemm, 0, stream>>>(XB1, S, W2, XB2, N);
    gemm_xs_bf16<8><<<gGemm, bGemm, 0, stream>>>(XB2, S, W3, XB3, N);
    combine2<<<(B * N + 255) / 256, 256, 0, stream>>>(Y0f, W1, W2, W3, H2, b2, Wro, bro, out, N);
}
